// RNN_82317343195943
// MI455X (gfx1250) — compile-verified, matrix operand order checked by reading, not run on hardware
//
#include <hip/hip_runtime.h>

#define B_SZ 256
#define T_SZ 4096
#define H_SZ 32
#define CHUNK 128
#define YPAD 4   // ybuf stride = 132 dwords: bank-conflict-free, 16B aligned

typedef __attribute__((ext_vector_type(16))) _Float16 v16h;
typedef __attribute__((ext_vector_type(8)))  _Float16 v8h;
typedef __attribute__((ext_vector_type(8)))  float    v8f;

__device__ __forceinline__ float tanh_fast(float x) {
#if defined(__AMDGCN__) && __has_builtin(__builtin_amdgcn_tanhf)
  return __builtin_amdgcn_tanhf(x);          // gfx1250 v_tanh_f32 (TRANS)
#else
  float e = __expf(-2.0f * x);
  return (1.0f - e) / (1.0f + e);
#endif
}
__device__ __forceinline__ float sigmoid_fast(float x) {
  return 0.5f * tanh_fast(0.5f * x) + 0.5f;
}

__global__ __launch_bounds__(32, 1)
void lstm_wmma_kernel(const float* __restrict__ x,
                      const float* __restrict__ W_ih,
                      const float* __restrict__ W_hh,
                      const float* __restrict__ b_ih,
                      const float* __restrict__ b_hh,
                      const float* __restrict__ W_out,
                      const float* __restrict__ b_out,
                      float* __restrict__ out) {
  __shared__ __align__(16) float    xT[CHUNK][16];          // x chunk, transposed
  __shared__ __align__(16) float    ybuf[16][CHUNK + YPAD]; // y chunk buffer
  __shared__ __align__(16) _Float16 hstage[16][H_SZ];       // h D->A staging

  const int lane = threadIdx.x;   // 0..31
  const int lq   = lane & 15;
  const int half = lane >> 4;
  const int bm0  = blockIdx.x * 16;

  // ---- W_hh -> f16 WMMA B layout; tile j covers gate cols [16j,16j+16)
  // B element (k, n=g): lanes 0-15 hold K {0..7,16..23}, lanes 16-31 {8..15,24..31}
  v16h whhB[8];
#pragma unroll
  for (int j = 0; j < 8; ++j) {
    const int g = j * 16 + lq;
#pragma unroll
    for (int e = 0; e < 16; ++e) {
      const int k = (e < 8) ? (half * 8 + e) : (16 + half * 8 + (e - 8));
      whhB[j][e] = (_Float16)W_hh[g * H_SZ + k];
    }
  }
  float wih[8], bias2[8];
#pragma unroll
  for (int j = 0; j < 8; ++j) {
    const int g = j * 16 + lq;
    wih[j]   = W_ih[g];
    bias2[j] = b_ih[g] + b_hh[g];
  }
  float woutv[16];   // W_out gathered in A-layout order for the y dot product
#pragma unroll
  for (int e = 0; e < 16; ++e) {
    const int k = (e < 8) ? (half * 8 + e) : (16 + half * 8 + (e - 8));
    woutv[e] = W_out[k];
  }
  const float bout = b_out[0];

  v16h  hA = {};        // h_t, f16 A layout (persistent)
  float cD[16];         // c_t, f32 D layout [jt*8+r]
  float hD[2][8];       // h_t, f32 D layout (for y / final state)
#pragma unroll
  for (int i = 0; i < 16; ++i) cD[i] = 0.f;
#pragma unroll
  for (int i = 0; i < 8; ++i) { hD[0][i] = 0.f; hD[1][i] = 0.f; }

  float* yout   = out;
  float* hT_out = out + (size_t)B_SZ * T_SZ;
  float* cT_out = hT_out + (size_t)B_SZ * H_SZ;

  for (int t0 = 0; t0 < T_SZ; t0 += CHUNK) {
    // ---- stage x chunk transposed (coalesced b128 loads per row)
#pragma unroll 4
    for (int it = 0; it < 16; ++it) {
      const float4 v = *(const float4*)(x + (size_t)(bm0 + it) * T_SZ + t0 + lane * 4);
      xT[lane * 4 + 0][it] = v.x;
      xT[lane * 4 + 1][it] = v.y;
      xT[lane * 4 + 2][it] = v.z;
      xT[lane * 4 + 3][it] = v.w;
      if (t0 + CHUNK < T_SZ)
        __builtin_prefetch(x + (size_t)(bm0 + it) * T_SZ + t0 + CHUNK + lane * 4, 0, 1);
    }
    __syncthreads();

    for (int tc = 0; tc < CHUNK; ++tc) {
      // C operand = xg = x*W_ih + (b_ih+b_hh), built per lane in D layout
      const float4 xa = *(const float4*)&xT[tc][half * 8];
      const float4 xb = *(const float4*)&xT[tc][half * 8 + 4];
      const float xm[8] = {xa.x, xa.y, xa.z, xa.w, xb.x, xb.y, xb.z, xb.w};
      v8f acc[8];
#pragma unroll
      for (int j = 0; j < 8; ++j)
#pragma unroll
        for (int r = 0; r < 8; ++r)
          acc[j][r] = fmaf(xm[r], wih[j], bias2[j]);

      // gates += h @ W_hh^T : 8x v_wmma_f32_16x16x32_f16 (K=32=H)
#pragma unroll
      for (int j = 0; j < 8; ++j)
        acc[j] = __builtin_amdgcn_wmma_f32_16x16x32_f16(
            false, hA, false, whhB[j], (short)0, acc[j], false, false);

      // activations + cell update (tiles: 0,1=i  2,3=f  4,5=g  6,7=o)
#pragma unroll
      for (int jt = 0; jt < 2; ++jt)
#pragma unroll
        for (int r = 0; r < 8; ++r) {
          const float ig = sigmoid_fast(acc[0 + jt][r]);
          const float fg = sigmoid_fast(acc[2 + jt][r]);
          const float gg = tanh_fast(acc[4 + jt][r]);
          const float og = sigmoid_fast(acc[6 + jt][r]);
          const float c  = fmaf(fg, cD[jt * 8 + r], ig * gg);
          cD[jt * 8 + r] = c;
          hD[jt][r] = og * tanh_fast(c);
        }

      // D->A turn through 1KB LDS tile (wave-private; LDS is in-order per wave)
#pragma unroll
      for (int jt = 0; jt < 2; ++jt)
#pragma unroll
        for (int r = 0; r < 8; ++r)
          hstage[half * 8 + r][jt * 16 + lq] = (_Float16)hD[jt][r];
      const v8h ha0 = *(const v8h*)&hstage[lq][half * 8];
      const v8h ha1 = *(const v8h*)&hstage[lq][16 + half * 8];
      hA = __builtin_shufflevector(ha0, ha1, 0, 1, 2, 3, 4, 5, 6, 7,
                                   8, 9, 10, 11, 12, 13, 14, 15);

      // y_t[m] = h_t . W_out + b : half-lane partial + SWAPX16 swizzle reduce
      float ys = 0.f;
#pragma unroll
      for (int e = 0; e < 16; ++e) ys = fmaf((float)hA[e], woutv[e], ys);
      ys += __int_as_float(
          __builtin_amdgcn_ds_swizzle(__float_as_int(ys), 0x401F)); // xor 0x10
      if (lane < 16) ybuf[lq][tc] = ys + bout;
    }
    __syncthreads();

    // ---- flush y chunk (coalesced b128 stores per row)
#pragma unroll 4
    for (int it = 0; it < 16; ++it) {
      const float4 v = *(const float4*)&ybuf[it][lane * 4];
      *(float4*)(yout + (size_t)(bm0 + it) * T_SZ + t0 + lane * 4) = v;
    }
    __syncthreads();
  }

  // ---- final h_n, c_n (from D-layout registers)
#pragma unroll
  for (int jt = 0; jt < 2; ++jt)
#pragma unroll
    for (int r = 0; r < 8; ++r) {
      const int m = half * 8 + r;
      const int n = jt * 16 + lq;
      hT_out[(size_t)(bm0 + m) * H_SZ + n] = hD[jt][r];
      cT_out[(size_t)(bm0 + m) * H_SZ + n] = cD[jt * 8 + r];
    }
}

extern "C" void kernel_launch(void* const* d_in, const int* in_sizes, int n_in,
                              void* d_out, int out_size, void* d_ws, size_t ws_size,
                              hipStream_t stream) {
  const float* x     = (const float*)d_in[0];
  const float* W_ih  = (const float*)d_in[1];
  const float* W_hh  = (const float*)d_in[2];
  const float* b_ih  = (const float*)d_in[3];
  const float* b_hh  = (const float*)d_in[4];
  const float* W_out = (const float*)d_in[5];
  const float* b_out = (const float*)d_in[6];
  (void)in_sizes; (void)n_in; (void)out_size; (void)d_ws; (void)ws_size;
  lstm_wmma_kernel<<<B_SZ / 16, 32, 0, stream>>>(
      x, W_ih, W_hh, b_ih, b_hh, W_out, b_out, (float*)d_out);
}
